// PrimaryNetwork_56289841381713
// MI455X (gfx1250) — matrix operand order repaired and checked by reading, compiled, not run on hardware
//
#include <hip/hip_runtime.h>
#include <hip/hip_bf16.h>
#include <math.h>

#define O_DIM   32
#define A_DIM   8
#define H_DIM   256
#define W_ENS   128
#define N_BATCH 512
#define NT      128                  // batch rows per workgroup
#define NTILES  (N_BATCH / NT)       // 4
#define W_TOTAL 348194
#define W_PAD   348200               // bf16 scratch row stride (mult of 8 -> 16B aligned frags)

// ---- weight row offsets (floats / bf16 elements) inside one member row ----
#define OFF_WOBS   0        // (H, O)
#define OFF_BOBS   8192
#define OFF_WACT   8448     // (H, A)
#define OFF_BACT   10496
#define OFF_WTGT   10752    // (H, 2H)
#define OFF_BTGT   141824
#define OFF_WS0    142080   // (H, H)
#define OFF_BS0    207616
#define OFF_WS1    207872
#define OFF_BS1    273408
#define OFF_WS2    273664
#define OFF_BS2    339200
#define OFF_WSTATE 339456   // (O, H)
#define OFF_BSTATE 347648
#define OFF_WREW   347680   // (1, H)
#define OFF_BREW   347936
#define OFF_WDONE  347937   // (1, H)
#define OFF_BDONE  348193

typedef __attribute__((ext_vector_type(16))) __bf16 v16bf;
typedef __attribute__((ext_vector_type(8)))  __bf16 v8bf;
typedef __attribute__((ext_vector_type(8)))  float  v8f;

// leaky-relu(0.01): slope < 1  =>  identical to max(v, 0.01*v); 2 dual-issueable
// VALU ops (v_mul + v_max_num_f32), no VCC writes.
__device__ __forceinline__ float lrelu(float v) { return fmaxf(v, 0.01f * v); }

// A fragment (16x32 bf16) from LDS, row-major activations, leading dim ld.
// lane<16 -> M=lane, K=k0+{0..15}; lane>=16 -> M=lane-16, K=k0+{16..31}
__device__ __forceinline__ v16bf lds_load_a(const __bf16* y, int ld, int m0, int k0, int lane) {
    int row = m0 + (lane & 15);
    int kb  = k0 + ((lane & 16) ? 16 : 0);
    const __bf16* p = y + row * ld + kb;
    v8bf lo = *(const v8bf*)p;
    v8bf hi = *(const v8bf*)(p + 8);
    return __builtin_shufflevector(lo, hi, 0,1,2,3,4,5,6,7,8,9,10,11,12,13,14,15);
}

// B fragment (32x16 bf16) from pre-converted bf16 weights: lane holds 16
// consecutive K for its column n0+(lane&15). Two aligned 16B loads.
__device__ __forceinline__ v16bf glb_load_b16(const __bf16* wmat, int ld, int n0, int k0, int lane) {
    int col = n0 + (lane & 15);
    int kb  = k0 + ((lane & 16) ? 16 : 0);
    const __bf16* p = wmat + (size_t)col * ld + kb;
    v8bf lo = *(const v8bf*)p;
    v8bf hi = *(const v8bf*)(p + 8);
    return __builtin_shufflevector(lo, hi, 0,1,2,3,4,5,6,7,8,9,10,11,12,13,14,15);
}

// Epilogue for a 64x32 super-tile (acc[8] = [s][j]): bias + leaky-relu + bf16 store.
__device__ __forceinline__ void store_tile2(__bf16* yout, int ldo, int m0, int n0, int lane,
                                            const v8f* acc, float bv0, float bv1, bool relu) {
    int nc = (lane & 15);
    int mb = (lane & 16) ? 8 : 0;
#pragma unroll
    for (int s = 0; s < 4; ++s) {
#pragma unroll
        for (int j = 0; j < 2; ++j) {
            v8f c = acc[s * 2 + j];
            float bv = j ? bv1 : bv0;
            int col = n0 + 16 * j + nc;
#pragma unroll
            for (int r = 0; r < 8; ++r) {
                float v = c[r] + bv;
                if (relu) v = lrelu(v);
                yout[(size_t)(m0 + 16 * s + mb + r) * ldo + col] = (__bf16)v;
            }
        }
    }
}

// Dense layer: out = lrelu(in @ W^T + b). W is (hout,kdim) bf16 row-major.
// Wave super-tile = 64 rows x 32 cols: A reused across 2 B frags, B across 4 A frags.
__device__ __forceinline__ void gemm_layer(const __bf16* yin, int ldi, __bf16* yout, int ldo,
                                           const __bf16* wmat, const float* bias,
                                           int hout, int kdim, int wave, int lane) {
    const int nblk  = hout >> 5;
    const int tiles = nblk * (NT >> 6);
    for (int t = wave; t < tiles; t += 8) {
        int n0 = (t % nblk) << 5;
        int m0 = (t / nblk) << 6;
        v8f acc[8] = {};
        for (int k0 = 0; k0 < kdim; k0 += 32) {
            v16bf b0 = glb_load_b16(wmat, kdim, n0,      k0, lane);
            v16bf b1 = glb_load_b16(wmat, kdim, n0 + 16, k0, lane);
#pragma unroll
            for (int s = 0; s < 4; ++s) {
                v16bf a = lds_load_a(yin, ldi, m0 + 16 * s, k0, lane);
                acc[2 * s]     = __builtin_amdgcn_wmma_f32_16x16x32_bf16(
                    false, a, false, b0, (short)0, acc[2 * s], false, false);
                acc[2 * s + 1] = __builtin_amdgcn_wmma_f32_16x16x32_bf16(
                    false, a, false, b1, (short)0, acc[2 * s + 1], false, false);
            }
        }
        float bv0 = bias[n0 + (lane & 15)];
        float bv1 = bias[n0 + 16 + (lane & 15)];
        store_tile2(yout, ldo, m0, n0, lane, acc, bv0, bv1, true);
    }
}

// ---- pre-pass: convert all ensemble weights f32 -> bf16 into scratch -------
extern "C" __global__ void __launch_bounds__(256)
wm_convert_kernel(const float* __restrict__ w, __bf16* __restrict__ wb) {
    const int total = W_ENS * W_TOTAL;
    for (int idx = blockIdx.x * 256 + threadIdx.x; idx < total;
         idx += (int)gridDim.x * 256) {
        int wi = idx / W_TOTAL;
        int r  = idx - wi * W_TOTAL;
        wb[(size_t)wi * W_PAD + r] = (__bf16)w[idx];
    }
}

extern "C" __global__ void __launch_bounds__(256)
world_model_kernel(const float* __restrict__ obs, const float* __restrict__ act,
                   const float* __restrict__ w, const __bf16* __restrict__ wbf,
                   const float* __restrict__ obs_min, const float* __restrict__ obs_max,
                   float* __restrict__ out) {
    extern __shared__ __align__(16) char smem[];
    __bf16* xin  = (__bf16*)smem;                       // NT x 64   (16 KB)
    __bf16* bufA = (__bf16*)(smem + 16384);             // NT x 512  (128 KB)
    __bf16* bufB = (__bf16*)(smem + 16384 + 131072);    // NT x 256  (64 KB)

    const int wi   = blockIdx.x >> 2;          // ensemble member
    const int n0b  = (blockIdx.x & 3) * NT;    // batch tile base
    const int tid  = threadIdx.x;
    const int lane = tid & 31;
    const int wave = tid >> 5;
    const float*  wp = w   + (size_t)wi * W_TOTAL;   // f32 (biases, heads, act)
    const __bf16* wb = wbf + (size_t)wi * W_PAD;     // bf16 (big matrices)

    // ---- stage 0: build padded input [x_norm(32) | action(8) | 0(24)]
    for (int i = tid; i < NT * 64; i += 256) {
        int nl = i >> 6, c = i & 63;
        int n = n0b + nl;
        float v = 0.f;
        if (c < O_DIM) {
            float mn = obs_min[c], mx = obs_max[c];
            v = 2.f * (obs[n * O_DIM + c] - mn) / (mx - mn) - 1.f;   // SCALE=1
        } else if (c < O_DIM + A_DIM) {
            v = act[n * A_DIM + (c - O_DIM)];
        }
        xin[i] = (__bf16)v;
    }
    __syncthreads();

    // ---- layer 0: h0 = lrelu([x@Wobs^T | a@Wact^T] + b) -> bufA (NT x 512)
    {
        const int nblk = 512 >> 5;              // 16 super-cols of 32
        const int tiles = nblk * (NT >> 6);     // 32
        for (int t = wave; t < tiles; t += 8) {
            int n0 = (t % nblk) << 5;
            int m0 = (t / nblk) << 6;
            bool isObs = (n0 < H_DIM);          // 256 boundary aligns to 32-block
            int k0 = isObs ? 0 : 32;            // xin column base for this half
            v16bf b0, b1;
            if (isObs) {
                b0 = glb_load_b16(wb + OFF_WOBS, O_DIM, n0,      0, lane);
                b1 = glb_load_b16(wb + OFF_WOBS, O_DIM, n0 + 16, 0, lane);
            } else {
#pragma unroll
                for (int j = 0; j < 2; ++j) {
                    int col = (n0 + 16 * j - H_DIM) + (lane & 15);
                    int kb  = (lane & 16) ? 16 : 0;
                    const float* p = wp + OFF_WACT + (size_t)col * A_DIM;
                    v16bf b;
#pragma unroll
                    for (int i = 0; i < 16; ++i) {
                        int k = kb + i;
                        b[i] = (k < A_DIM) ? (__bf16)p[k] : (__bf16)0.0f;
                    }
                    if (j == 0) b0 = b; else b1 = b;
                }
            }
            v8f acc[8] = {};
#pragma unroll
            for (int s = 0; s < 4; ++s) {
                v16bf a = lds_load_a(xin, 64, m0 + 16 * s, k0, lane);
                acc[2 * s]     = __builtin_amdgcn_wmma_f32_16x16x32_bf16(
                    false, a, false, b0, (short)0, acc[2 * s], false, false);
                acc[2 * s + 1] = __builtin_amdgcn_wmma_f32_16x16x32_bf16(
                    false, a, false, b1, (short)0, acc[2 * s + 1], false, false);
            }
            const float* bp = isObs ? (wp + OFF_BOBS + n0) : (wp + OFF_BACT + n0 - H_DIM);
            store_tile2(bufA, 512, m0, n0, lane, acc,
                        bp[lane & 15], bp[16 + (lane & 15)], true);
        }
    }
    __syncthreads();

    // ---- tgt layer (K=512) and three hidden layers (K=256), ping-pong LDS
    gemm_layer(bufA, 512, bufB, 256, wb + OFF_WTGT, wp + OFF_BTGT, H_DIM, 512, wave, lane);
    __syncthreads();
    gemm_layer(bufB, 256, bufA, 256, wb + OFF_WS0, wp + OFF_BS0, H_DIM, 256, wave, lane);
    __syncthreads();
    gemm_layer(bufA, 256, bufB, 256, wb + OFF_WS1, wp + OFF_BS1, H_DIM, 256, wave, lane);
    __syncthreads();
    gemm_layer(bufB, 256, bufA, 256, wb + OFF_WS2, wp + OFF_BS2, H_DIM, 256, wave, lane);
    __syncthreads();
    // final activations y in bufA (NT x 256)

    // ---- state head: 64x32 super-tiles over K=256, fused residual + denorm
    for (int t = wave; t < 2; t += 8) {          // waves 0..1
        int m0 = t << 6;
        v8f acc[8] = {};
        for (int k0 = 0; k0 < H_DIM; k0 += 32) {
            v16bf b0 = glb_load_b16(wb + OFF_WSTATE, H_DIM, 0,  k0, lane);
            v16bf b1 = glb_load_b16(wb + OFF_WSTATE, H_DIM, 16, k0, lane);
#pragma unroll
            for (int s = 0; s < 4; ++s) {
                v16bf a = lds_load_a(bufA, 256, m0 + 16 * s, k0, lane);
                acc[2 * s]     = __builtin_amdgcn_wmma_f32_16x16x32_bf16(
                    false, a, false, b0, (short)0, acc[2 * s], false, false);
                acc[2 * s + 1] = __builtin_amdgcn_wmma_f32_16x16x32_bf16(
                    false, a, false, b1, (short)0, acc[2 * s + 1], false, false);
            }
        }
        int mb = (lane & 16) ? 8 : 0;
#pragma unroll
        for (int j = 0; j < 2; ++j) {
            int o = 16 * j + (lane & 15);
            float mn  = obs_min[o], mx = obs_max[o];
            float rng = mx - mn;
            float bst = wp[OFF_BSTATE + o];
#pragma unroll
            for (int s = 0; s < 4; ++s) {
                v8f c = acc[2 * s + j];
#pragma unroll
                for (int r = 0; r < 8; ++r) {
                    int n = n0b + m0 + 16 * s + mb + r;
                    float xn = 2.f * (obs[n * O_DIM + o] - mn) / rng - 1.f;
                    float v  = c[r] + bst + xn;              // + residual
                    v = (v + 1.f) * 0.5f * rng + mn;         // de-normalize
                    out[(size_t)n * (W_ENS * O_DIM) + (size_t)wi * O_DIM + o] = v;
                }
            }
        }
    }

    // ---- reward / done heads: 256-length dot per row (VALU), tanh / sigmoid
    {
        int nl    = tid & 127;
        int which = tid >> 7;                    // 0 = reward, 1 = done
        const __bf16* yr = bufA + (size_t)nl * 256;
        const float* wv  = wp + (which ? OFF_WDONE : OFF_WREW);
        float bsc        = wp[which ? OFF_BDONE : OFF_BREW];
        float d = 0.f;
#pragma unroll 8
        for (int k = 0; k < H_DIM; ++k) d += (float)yr[k] * wv[k];
        float z = d + bsc;
        float r = which ? (1.f / (1.f + expf(-z))) : tanhf(z);
        int n = n0b + nl;
        out[(size_t)N_BATCH * W_ENS * O_DIM + (size_t)which * N_BATCH * W_ENS
            + (size_t)n * W_ENS + wi] = r;
    }
}

extern "C" void kernel_launch(void* const* d_in, const int* in_sizes, int n_in,
                              void* d_out, int out_size, void* d_ws, size_t ws_size,
                              hipStream_t stream) {
    (void)in_sizes; (void)n_in; (void)out_size; (void)ws_size;
    const float* obs     = (const float*)d_in[0];
    const float* act     = (const float*)d_in[1];
    const float* w       = (const float*)d_in[2];
    const float* obs_min = (const float*)d_in[3];
    const float* obs_max = (const float*)d_in[4];
    float*  out = (float*)d_out;
    __bf16* wbf = (__bf16*)d_ws;     // 128 * 348200 * 2B = 89.1 MB scratch

    wm_convert_kernel<<<dim3(1024), dim3(256), 0, stream>>>(w, wbf);

    constexpr size_t SMEM = 16384 + 131072 + 65536;   // 208 KB dynamic LDS
    hipFuncSetAttribute((const void*)world_model_kernel,
                        hipFuncAttributeMaxDynamicSharedMemorySize, (int)SMEM);
    world_model_kernel<<<dim3(W_ENS * NTILES), dim3(256), SMEM, stream>>>(
        obs, act, w, wbf, obs_min, obs_max, out);
}
